// MGSTGCNwmt_44779329028460
// MI455X (gfx1250) — matrix operand order repaired and checked by reading, not compiled, not run on hardware
//
#include <hip/hip_runtime.h>

// ---------------------------------------------------------------------------
// MGSTGCN forward for gfx1250 (MI455X).  All GEMM-shaped ops go through one
// generic im2col WMMA kernel using v_wmma_f32_16x16x32_f16 (f16 in, f32 acc),
// templated on Kt / W-transpose / channel-split; staging uses wave-uniform
// fast paths (readfirstlane) so in-bounds tiles do unpredicated loads.
// ---------------------------------------------------------------------------

typedef __attribute__((ext_vector_type(16))) _Float16 v16h;
typedef __attribute__((ext_vector_type(8)))  _Float16 v8h;
typedef __attribute__((ext_vector_type(4)))  _Float16 v4h;
typedef __attribute__((ext_vector_type(8)))  float    v8f;

#define GRID1(n) dim3((unsigned int)(((n) + 255) / 256))

__device__ __forceinline__ float sigf(float x) { return 1.f / (1.f + __expf(-x)); }

__device__ __forceinline__ bool uniform_bool(bool c)
{
    return __builtin_amdgcn_readfirstlane((int)c) != 0;
}

// ---------------------------------------------------------------------------
// Generic WMMA GEMM:  D[b] = W[b] (O x K) @ X[b] (K x ncols)  (+bias, epilogue)
//   K = Cin*KT ; k -> (cin = k/KT, j = k%KT) ; col -> (n = col/Lout, t = col%Lout)
//   X element addr = b*xBatch + cin*xChan + n*xRow + t + j*dil
//   (SPLIT: cin<csplit -> X0 else X1) ; WT: W stored transposed (k*O + o)
//   Out addr       = b*oBatch + o*oChan + n*oRow + t
//   mode: 0 none, 1 relu, 2 sigmoid(v)*Aux, 3 accumulate into Out
//   NOTE: all call sites have K % 4 == 0 (float4 weight loads rely on this).
// ---------------------------------------------------------------------------
template<int KT, bool WT, bool SPLIT>
__global__ __launch_bounds__(128) void conv_gemm_wmma(
    const float* __restrict__ Wm, long long wBatch,
    const float* __restrict__ bias,
    const float* __restrict__ X0, const float* __restrict__ X1, int csplit,
    float* __restrict__ Outp, const float* __restrict__ Aux,
    int O, int Cin, int dil, int ncols, int Lout,
    long long xBatch, int xChan, int xRow,
    long long oBatch, int oChan, int oRow,
    long long aBatch, int aChan, int aRow,
    int mode, int inRelu)
{
    __shared__ alignas(16) _Float16 As[16][32];
    __shared__ alignas(16) _Float16 Bs[4][16][32];

    const int tid  = threadIdx.x;
    const int wave = tid >> 5;
    const int lane = tid & 31;
    const int b    = blockIdx.z;
    const int oTile = blockIdx.y;
    const int myColTile = blockIdx.x * 4 + wave;
    const int K = Cin * KT;
    const float* Wb = Wm + (long long)b * wBatch;

    // hoisted A-staging coords: 4 consecutive k per thread, fixed row
    const int arow = tid >> 3;
    const int akk  = (tid & 7) * 4;
    const int aO   = oTile * 16 + arow;
    const bool aok = (aO < O);
    const bool oTileFull = uniform_bool(oTile * 16 + 16 <= O);

    // hoisted B-staging coords: 16 consecutive k per lane, fixed column
    const int bcl  = lane >> 1;
    const int bkk  = (lane & 1) * 16;
    const int bcol = myColTile * 16 + bcl;
    const bool bok = (bcol < ncols);
    const bool colsFull = uniform_bool(myColTile * 16 + 16 <= ncols);
    int bn = 0, bt = 0;
    if (bok) { bn = bcol / Lout; bt = bcol - bn * Lout; }
    const long long xcolbase = (long long)b * xBatch + (long long)bn * xRow + bt;

    v8f acc = {0.f, 0.f, 0.f, 0.f, 0.f, 0.f, 0.f, 0.f};
    const int nchunks = (K + 31) >> 5;

    for (int kc = 0; kc < nchunks; ++kc) {
        const int kbase = kc << 5;
        const bool fullK = (kbase + 32 <= K);   // scalar (K, kbase uniform)

        // ---- stage W tile (shared across the 4 waves) ----
        {
            const int kg = kbase + akk;
            v4h p;
            if (oTileFull && fullK) {           // uniform -> unpredicated loads
                if (!WT) {
                    const float4 w4 = *(const float4*)(Wb + (long long)aO * K + kg);
                    p[0] = (_Float16)w4.x; p[1] = (_Float16)w4.y;
                    p[2] = (_Float16)w4.z; p[3] = (_Float16)w4.w;
                } else {
#pragma unroll
                    for (int i = 0; i < 4; ++i)
                        p[i] = (_Float16)Wb[(long long)(kg + i) * O + aO];
                }
            } else {
#pragma unroll
                for (int i = 0; i < 4; ++i) {
                    int k = kg + i;
                    float v = 0.f;
                    if (aok && k < K)
                        v = WT ? Wb[(long long)k * O + aO] : Wb[(long long)aO * K + k];
                    p[i] = (_Float16)v;
                }
            }
            *(v4h*)&As[arow][akk] = p;
        }
        // ---- stage X tile per wave, transposed: Bs[col_local][k] ----
        {
            v8h p0, p1;
            if (colsFull && fullK) {            // uniform -> unpredicated loads
#pragma unroll
                for (int i = 0; i < 16; ++i) {
                    const int k = kbase + bkk + i;
                    int cin, j;
                    if (KT == 1) { cin = k; j = 0; }
                    else         { cin = k / KT; j = k - cin * KT; }
                    const float* src = X0; int cc = cin;
                    if (SPLIT && cin >= csplit) { src = X1; cc = cin - csplit; }
                    float v = src[xcolbase + (long long)cc * xChan + j * dil];
                    if (inRelu) v = v > 0.f ? v : 0.f;
                    if (i < 8) p0[i] = (_Float16)v; else p1[i - 8] = (_Float16)v;
                }
            } else {
#pragma unroll
                for (int i = 0; i < 16; ++i) {
                    const int k = kbase + bkk + i;
                    float v = 0.f;
                    if (bok && k < K) {
                        int cin, j;
                        if (KT == 1) { cin = k; j = 0; }
                        else         { cin = k / KT; j = k - cin * KT; }
                        const float* src = X0; int cc = cin;
                        if (SPLIT && cin >= csplit) { src = X1; cc = cin - csplit; }
                        v = src[xcolbase + (long long)cc * xChan + j * dil];
                        if (inRelu) v = v > 0.f ? v : 0.f;
                    }
                    if (i < 8) p0[i] = (_Float16)v; else p1[i - 8] = (_Float16)v;
                }
            }
            *(v8h*)&Bs[wave][bcl][bkk]     = p0;
            *(v8h*)&Bs[wave][bcl][bkk + 8] = p1;
        }
        __syncthreads();
        // ---- fragments per CDNA5 16-bit A (16x32) / B (32x16) VGPR layouts ----
        {
            const int m = lane & 15, hi = lane >> 4;
            v8h alo = *(const v8h*)&As[m][hi * 8];
            v8h ahi = *(const v8h*)&As[m][16 + hi * 8];
            v8h blo = *(const v8h*)&Bs[wave][m][hi * 8];
            v8h bhi = *(const v8h*)&Bs[wave][m][16 + hi * 8];
            v16h a, bb;
#pragma unroll
            for (int i = 0; i < 8; ++i) {
                a[i] = alo[i];  a[i + 8] = ahi[i];
                bb[i] = blo[i]; bb[i + 8] = bhi[i];
            }
            acc = __builtin_amdgcn_wmma_f32_16x16x32_f16(false, a, false, bb,
                                                         (short)0, acc, false, false);
        }
        __syncthreads();
    }

    // epilogue: acc[r] = D[o = oTile*16 + r + 8*hi, col = myColTile*16 + (lane&15)]
    const int hi = lane >> 4;
    const int colG = myColTile * 16 + (lane & 15);
    if (colG < ncols) {
        const int n = colG / Lout, t = colG - n * Lout;
#pragma unroll
        for (int r = 0; r < 8; ++r) {
            const int o = oTile * 16 + r + 8 * hi;
            if (o < O) {
                float v = acc[r] + (bias ? bias[o] : 0.f);
                const long long oaddr = (long long)b * oBatch + (long long)o * oChan
                                      + (long long)n * oRow + t;
                if (mode == 1) {
                    v = v > 0.f ? v : 0.f;
                } else if (mode == 2) {
                    const long long aaddr = (long long)b * aBatch + (long long)o * aChan
                                          + (long long)n * aRow + t;
                    v = sigf(v) * Aux[aaddr];
                } else if (mode == 3) {
                    v += Outp[oaddr];
                }
                Outp[oaddr] = v;
            }
        }
    }
}

// ---------------------------------------------------------------------------
// Elementwise / reduction kernels
// ---------------------------------------------------------------------------

// per-channel mean / rsqrt(var+eps) over (B, inner); X layout (B,Ch,inner)
__global__ void bn_stats_kernel(const float* __restrict__ X, float* __restrict__ mv,
                                int Bn, int Ch, long long inner)
{
    int c = blockIdx.x;
    __shared__ float ss[256], sq[256];
    float s = 0.f, q = 0.f;
    long long tot = (long long)Bn * inner;
    for (long long i = threadIdx.x; i < tot; i += 256) {
        long long b = i / inner, r = i - b * inner;
        float v = X[(b * Ch + c) * inner + r];
        s += v; q += v * v;
    }
    ss[threadIdx.x] = s; sq[threadIdx.x] = q; __syncthreads();
    for (int st = 128; st > 0; st >>= 1) {
        if (threadIdx.x < st) { ss[threadIdx.x] += ss[threadIdx.x + st]; sq[threadIdx.x] += sq[threadIdx.x + st]; }
        __syncthreads();
    }
    if (threadIdx.x == 0) {
        float cnt = (float)tot;
        float m = ss[0] / cnt;
        float var = sq[0] / cnt - m * m;
        mv[2 * c] = m; mv[2 * c + 1] = rsqrtf(var + 1e-5f);
    }
}

__global__ void bn_apply_kernel(const float* __restrict__ X, float* __restrict__ Y,
                                const float* __restrict__ mv, int Ch, long long inner,
                                long long total)
{
    long long i = (long long)blockIdx.x * 256 + threadIdx.x;
    if (i >= total) return;
    int c = (int)((i / inner) % Ch);
    Y[i] = (X[i] - mv[2 * c]) * mv[2 * c + 1];
}

// fused input-BN + start 1x1 conv  (Cin <= 3, O = 32)
__global__ void start_conv_kernel(const float* __restrict__ X, const float* __restrict__ mv,
                                  const float* __restrict__ Wt, const float* __restrict__ bias,
                                  float* __restrict__ Y, int Cin, long long inner, long long total)
{
    long long i = (long long)blockIdx.x * 256 + threadIdx.x;
    if (i >= total) return;
    long long b = i / (32 * inner);
    long long r = i - b * 32 * inner;
    int o = (int)(r / inner);
    long long s = r - (long long)o * inner;
    float acc = bias[o];
    for (int c = 0; c < Cin; ++c) {
        float v = (X[(b * Cin + c) * inner + s] - mv[2 * c]) * mv[2 * c + 1];
        acc += Wt[o * Cin + c] * v;
    }
    Y[i] = acc;
}

// adaptive adjacency: A = rownorm(relu(v1 @ v2)), v1 (Nn,10), v2 (10,Nn)
__global__ void adaptive_kernel(const float* __restrict__ v1, const float* __restrict__ v2,
                                float* __restrict__ A, int Nn)
{
    int i = blockIdx.x;
    __shared__ float sred[256];
    float d[10];
#pragma unroll
    for (int k = 0; k < 10; ++k) d[k] = v1[i * 10 + k];
    float s = 0.f;
    for (int j = threadIdx.x; j < Nn; j += 256) {
        float acc = 0.f;
#pragma unroll
        for (int k = 0; k < 10; ++k) acc += d[k] * v2[k * Nn + j];
        acc = acc > 0.f ? acc : 0.f;
        A[i * Nn + j] = acc;
        s += acc;
    }
    sred[threadIdx.x] = s; __syncthreads();
    for (int st = 128; st > 0; st >>= 1) {
        if (threadIdx.x < st) sred[threadIdx.x] += sred[threadIdx.x + st];
        __syncthreads();
    }
    float inv = 1.f / sred[0];
    for (int j = threadIdx.x; j < Nn; j += 256) A[i * Nn + j] *= inv;
}

// ---- transmit_fn pieces (x always (B,32,512,L)) ----
__global__ void tr_f1_kernel(const float* __restrict__ X, const float* __restrict__ c1,
                             float* __restrict__ F1, int L, long long total)
{
    long long i = (long long)blockIdx.x * 256 + threadIdx.x;
    if (i >= total) return;
    long long b = i / (512LL * L);
    long long r = i - b * 512LL * L;      // n*L + l
    float acc = 0.f;
    for (int c = 0; c < 32; ++c) acc += X[(b * 32 + c) * 512LL * L + r] * c1[c];
    F1[i] = acc;
}

__global__ void tr_f2_kernel(const float* __restrict__ Xc, const float* __restrict__ c2,
                             float* __restrict__ F2, int M, int L, long long total)
{
    long long i = (long long)blockIdx.x * 256 + threadIdx.x;
    if (i >= total) return;                // (b,c,m)
    long long base = i * L;
    float acc = 0.f;
    for (int l = 0; l < L; ++l) acc += Xc[base + l] * c2[l];
    F2[i] = acc;
}

__global__ void tr_g_kernel(const float* __restrict__ F1, const float* __restrict__ w,
                            float* __restrict__ G, int L, long long total)
{
    long long i = (long long)blockIdx.x * 256 + threadIdx.x;
    if (i >= total) return;                // (b,n,c)
    long long bn = i / 32;
    int c = (int)(i - bn * 32);
    float acc = 0.f;
    for (int l = 0; l < L; ++l) acc += F1[bn * L + l] * w[l * 32 + c];
    G[i] = acc;
}

__global__ void tr_logits_kernel(const float* __restrict__ G, const float* __restrict__ F2,
                                 const float* __restrict__ tb, float* __restrict__ Lg,
                                 int M, long long total)
{
    long long i = (long long)blockIdx.x * 256 + threadIdx.x;
    if (i >= total) return;                // (b,n,m)
    long long b = i / (512LL * M);
    long long r = i - b * 512LL * M;
    long long n = r / M;
    int m = (int)(r - n * M);
    float acc = tb[n * M + m];
    for (int c = 0; c < 32; ++c)
        acc += G[(b * 512 + n) * 32 + c] * F2[(b * 32 + c) * M + m];
    Lg[i] = sigf(acc);
}

__global__ void tr_colmean_kernel(const float* __restrict__ Lg, float* __restrict__ Mean, int M)
{
    int bm = blockIdx.x;
    int b = bm / M, m = bm - b * M;
    __shared__ float ss[128];
    float s = 0.f;
    for (int n = threadIdx.x; n < 512; n += 128) s += Lg[((long long)b * 512 + n) * M + m];
    ss[threadIdx.x] = s; __syncthreads();
    for (int st = 64; st > 0; st >>= 1) {
        if (threadIdx.x < st) ss[threadIdx.x] += ss[threadIdx.x + st];
        __syncthreads();
    }
    if (threadIdx.x == 0) Mean[bm] = ss[0] / 512.f;
}

__global__ void tr_final_kernel(const float* __restrict__ Lg, const float* __restrict__ Mean,
                                const float* __restrict__ mask, float* __restrict__ T,
                                int M, long long total)
{
    long long i = (long long)blockIdx.x * 256 + threadIdx.x;
    if (i >= total) return;                // (b,n,m)
    long long b = i / (512LL * M);
    long long r = i - b * 512LL * M;
    long long n = r / M;
    int m = (int)(r - n * M);
    T[i] = sigf(Lg[i] - Mean[b * M + m]) * mask[n * M + m];
}

// copy x (B,32,inner) into channel-slot of a (B,96,inner) tensor
__global__ void copy_slot_kernel(const float* __restrict__ X, float* __restrict__ Y,
                                 long long perB, long long yB, long long total)
{
    long long i = (long long)blockIdx.x * 256 + threadIdx.x;
    if (i >= total) return;
    long long b = i / perB, r = i - b * perB;
    Y[b * yB + r] = X[i];
}

// GLU: H[b,c,s] = tanh(Y[b,c,s]) * sigmoid(Y[b,c+32,s])
__global__ void glu_kernel(const float* __restrict__ Y, float* __restrict__ H,
                           long long cinner, long long yB, long long hB, long long total)
{
    long long i = (long long)blockIdx.x * 256 + threadIdx.x;
    if (i >= total) return;
    long long b = i / (32 * cinner);
    long long r = i - b * 32 * cinner;
    H[b * hB + r] = tanhf(Y[b * yB + r]) * sigf(Y[b * yB + 32 * cinner + r]);
}

// ---- temporal attention pieces ; G layout (B,32,Nn,T), batch stride 32*Nn*T ----
__global__ void at_f1_kernel(const float* __restrict__ G, const float* __restrict__ c1,
                             float* __restrict__ F1, int Nn, int T, long long total)
{
    long long i = (long long)blockIdx.x * 256 + threadIdx.x;
    if (i >= total) return;                // (b,l,n)
    long long b = i / ((long long)T * Nn);
    long long r = i - b * T * Nn;
    int l = (int)(r / Nn);
    int n = (int)(r - (long long)l * Nn);
    float acc = 0.f;
    for (int c = 0; c < 32; ++c)
        acc += G[((b * 32 + c) * Nn + n) * T + l] * c1[c];
    F1[i] = acc;
}

__global__ void at_f2_kernel(const float* __restrict__ G, const float* __restrict__ c2,
                             float* __restrict__ F2, int Nn, int T, long long total)
{
    long long i = (long long)blockIdx.x * 256 + threadIdx.x;
    if (i >= total) return;                // (b,c,l)
    long long bc = i / T;
    int l = (int)(i - bc * T);
    float acc = 0.f;
    for (int n = 0; n < Nn; ++n) acc += G[(bc * Nn + n) * T + l] * c2[n];
    F2[i] = acc;
}

__global__ void at_mid_kernel(const float* __restrict__ F1, const float* __restrict__ w,
                              float* __restrict__ Mid, int Nn, int T, long long total)
{
    long long i = (long long)blockIdx.x * 256 + threadIdx.x;
    if (i >= total) return;                // (b,l,c)
    long long bl = i / 32;
    int c = (int)(i - bl * 32);
    float acc = 0.f;
    for (int n = 0; n < Nn; ++n) acc += F1[bl * Nn + n] * w[n * 32 + c];
    Mid[i] = acc;
}

__global__ void at_logits_kernel(const float* __restrict__ Mid, const float* __restrict__ F2,
                                 const float* __restrict__ tb, float* __restrict__ Lg,
                                 int T, long long total)
{
    long long i = (long long)blockIdx.x * 256 + threadIdx.x;
    if (i >= total) return;                // (b,l,q)
    long long b = i / ((long long)T * T);
    long long r = i - b * T * T;
    int l = (int)(r / T);
    int q = (int)(r - (long long)l * T);
    float acc = tb[l * T + q];
    for (int c = 0; c < 32; ++c)
        acc += Mid[(b * T + l) * 32 + c] * F2[(b * 32 + c) * T + q];
    Lg[i] = sigf(acc);
}

__global__ void at_v_kernel(const float* __restrict__ v, const float* __restrict__ Lg,
                            float* __restrict__ LV, int T, long long total)
{
    long long i = (long long)blockIdx.x * 256 + threadIdx.x;
    if (i >= total) return;                // (b,l,t)
    long long b = i / ((long long)T * T);
    long long r = i - b * T * T;
    int l = (int)(r / T);
    int t = (int)(r - (long long)l * T);
    float acc = 0.f;
    for (int q = 0; q < T; ++q) acc += v[l * T + q] * Lg[(b * T + q) * T + t];
    LV[i] = acc;
}

__global__ void at_bnstats_kernel(const float* __restrict__ LV, float* __restrict__ mv,
                                  int Bn, int T)
{
    int j = blockIdx.x;
    __shared__ float ss[128], sq[128];
    float s = 0.f, q = 0.f;
    for (int k = threadIdx.x; k < Bn * T; k += 128) {
        int b = k / T, i = k - b * T;
        float v = LV[((long long)b * T + i) * T + j];
        s += v; q += v * v;
    }
    ss[threadIdx.x] = s; sq[threadIdx.x] = q; __syncthreads();
    for (int st = 64; st > 0; st >>= 1) {
        if (threadIdx.x < st) { ss[threadIdx.x] += ss[threadIdx.x + st]; sq[threadIdx.x] += sq[threadIdx.x + st]; }
        __syncthreads();
    }
    if (threadIdx.x == 0) {
        float cnt = (float)(Bn * T);
        float m = ss[0] / cnt;
        float var = sq[0] / cnt - m * m;
        mv[2 * j] = m; mv[2 * j + 1] = rsqrtf(var + 1e-5f);
    }
}

__global__ void at_softmax_kernel(const float* __restrict__ LV, const float* __restrict__ mv,
                                  float* __restrict__ Coef, int T, long long total)
{
    long long i = (long long)blockIdx.x * 256 + threadIdx.x;
    if (i >= total) return;                // (b,i) rows
    float vv[8];
    float mx = -1e30f;
    for (int j = 0; j < T; ++j) {
        float v = (LV[i * T + j] - mv[2 * j]) * mv[2 * j + 1];
        vv[j] = v;
        mx = v > mx ? v : mx;
    }
    float sum = 0.f;
    for (int j = 0; j < T; ++j) { vv[j] = __expf(vv[j] - mx); sum += vv[j]; }
    float inv = 1.f / sum;
    for (int j = 0; j < T; ++j) Coef[i * T + j] = vv[j] * inv;
}

// out[b,c,n,q] = sum_l G[b,c,n,l] * Coef[b,q,l] + RES[b,c,n,q]
__global__ void at_apply_kernel(const float* __restrict__ G, const float* __restrict__ Coef,
                                const float* __restrict__ RES, float* __restrict__ Outp,
                                int Nn, int T, long long total)
{
    long long i = (long long)blockIdx.x * 256 + threadIdx.x;
    if (i >= total) return;
    long long NT = (long long)Nn * T;
    long long b = i / (32 * NT);
    long long r = i - b * 32 * NT;
    long long c = r / NT;
    long long s = r - c * NT;
    long long n = s / T;
    int q = (int)(s - n * T);
    float acc = RES[i];
    long long row = (b * 32 + c) * NT + n * T;
    for (int l = 0; l < T; ++l)
        acc += G[row + l] * Coef[(b * T + q) * T + l];
    Outp[i] = acc;
}

// ---------------------------------------------------------------------------
// Host-side helpers
// ---------------------------------------------------------------------------
static inline void gemm(hipStream_t st, const float* Wp, long long wB, int wT,
                        const float* bias, const float* X0, const float* X1, int csplit,
                        float* Outp, const float* Aux,
                        int O, int Cin, int Kt, int dil, int ncols, int Lout,
                        long long xB, int xC, int xR,
                        long long oB, int oC, int oR,
                        long long aB, int aC, int aR,
                        int mode, int inRelu, int batches)
{
    dim3 g((ncols + 63) / 64, (O + 15) / 16, batches);
    const bool split = (csplit < Cin);
#define GEMM_LAUNCH(KT_, WT_, SP_)                                                     \
    conv_gemm_wmma<KT_, WT_, SP_><<<g, dim3(128), 0, st>>>(Wp, wB, bias, X0, X1,       \
        csplit, Outp, Aux, O, Cin, dil, ncols, Lout, xB, xC, xR, oB, oC, oR,           \
        aB, aC, aR, mode, inRelu)
    if (Kt == 1) {
        if (wT)        GEMM_LAUNCH(1, true, false);
        else if (split) GEMM_LAUNCH(1, false, true);
        else            GEMM_LAUNCH(1, false, false);
    } else if (Kt == 2) {
        GEMM_LAUNCH(2, false, false);
    } else {
        GEMM_LAUNCH(3, false, false);
    }
#undef GEMM_LAUNCH
}

static void run_transmit(hipStream_t st, const float* x, const float* xc, int M, int Lc,
                         const float* c1, const float* c2, const float* w, const float* tb,
                         const float* mask, float* f1, float* f2, float* g, float* logits,
                         float* mean, float* tout)
{
    long long n1 = 64LL * 512 * Lc;
    tr_f1_kernel<<<GRID1(n1), 256, 0, st>>>(x, c1, f1, Lc, n1);
    long long n2 = 64LL * 32 * M;
    tr_f2_kernel<<<GRID1(n2), 256, 0, st>>>(xc, c2, f2, M, Lc, n2);
    long long n3 = 64LL * 512 * 32;
    tr_g_kernel<<<GRID1(n3), 256, 0, st>>>(f1, w, g, Lc, n3);
    long long n4 = 64LL * 512 * M;
    tr_logits_kernel<<<GRID1(n4), 256, 0, st>>>(g, f2, tb, logits, M, n4);
    tr_colmean_kernel<<<dim3(64 * M), dim3(128), 0, st>>>(logits, mean, M);
    tr_final_kernel<<<GRID1(n4), 256, 0, st>>>(logits, mean, mask, tout, M, n4);
}

static void run_gate(hipStream_t st, const float* x, const float* xa, const float* xb,
                     const float* const* P, int gb, float* xg, int Lc)
{
    long long NL = 512LL * Lc;
    copy_slot_kernel<<<GRID1(64LL * 32 * NL), 256, 0, st>>>(x, xg, 32 * NL, 96 * NL, 64LL * 32 * NL);
    gemm(st, P[gb + 0], 0, 0, P[gb + 1], xa, x, 32, xg + 32 * NL, xa,
         32, 64, 1, 1, (int)NL, Lc, 32 * NL, (int)NL, Lc,
         96 * NL, (int)NL, Lc, 32 * NL, (int)NL, Lc, 2, 0, 64);
    gemm(st, P[gb + 2], 0, 0, P[gb + 3], xb, x, 32, xg + 64 * NL, xb,
         32, 64, 1, 1, (int)NL, Lc, 32 * NL, (int)NL, Lc,
         96 * NL, (int)NL, Lc, 32 * NL, (int)NL, Lc, 2, 0, 64);
}

static void run_gcn_pool(hipStream_t st, const float* x, int cin, int Nn, int Lin,
                         const float* A, const float* const* P, int pb, int Kt,
                         float* outp, float* scr, float* sm)
{
    const long long Bn = 64;
    int Lh = Lin - 2 * (Kt - 1);
    int T  = Lh - (Kt - 1);
    long long NLin = (long long)Nn * Lin, NLh = (long long)Nn * Lh, NT = (long long)Nn * T;
    long long resSz = Bn * 32 * NT, ySz = Bn * 64 * NLh, hSz = Bn * 96 * NLh;

    float* res  = scr;
    float* y    = res + resSz;
    float* hcat = y + ySz;
    float* y2   = hcat + hSz;
    float* g    = y;                  // reuse (y dead after GLU -> hcat)
    float* tmp  = y + Bn * 32 * NT;   // reuse
    float* f1   = y2;                 // attention smalls reuse y2 (dead after GLU2)
    float* f2   = f1 + Bn * (long long)T * Nn;
    float* mid  = f2 + Bn * 32LL * T;
    float* lg   = mid + Bn * (long long)T * 32;
    float* lv   = lg + Bn * (long long)T * T;
    float* coef = lv + Bn * (long long)T * T;

    // residual 1x1 conv on last-T slice
    gemm(st, P[pb + 0], 0, 0, P[pb + 1], x + (Lin - T), x, 1 << 30, res, 0,
         32, cin, 1, 1, (int)NT, T, (long long)cin * NLin, (int)NLin, Lin,
         32 * NT, (int)NT, T, 0, 0, 0, 0, 0, 64);
    // dilated temporal conv (GLU pre-activation), dil=2
    gemm(st, P[pb + 2], 0, 0, P[pb + 3], x, x, 1 << 30, y, 0,
         64, cin, Kt, 2, (int)NLh, Lh, (long long)cin * NLin, (int)NLin, Lin,
         64 * NLh, (int)NLh, Lh, 0, 0, 0, 0, 0, 64);
    glu_kernel<<<GRID1(Bn * 32 * NLh), 256, 0, st>>>(y, hcat, NLh, 64 * NLh, 96 * NLh, Bn * 32 * NLh);
    // two graph-diffusion steps: W = A^T (wTrans)
    gemm(st, A, 0, 1, 0, hcat, hcat, 1 << 30, hcat + 32 * NLh, 0,
         Nn, Nn, 1, 1, 32 * Lh, Lh, 96 * NLh, Lh, (int)NLh,
         96 * NLh, Lh, (int)NLh, 0, 0, 0, 0, 0, 64);
    gemm(st, A, 0, 1, 0, hcat + 32 * NLh, hcat, 1 << 30, hcat + 64 * NLh, 0,
         Nn, Nn, 1, 1, 32 * Lh, Lh, 96 * NLh, Lh, (int)NLh,
         96 * NLh, Lh, (int)NLh, 0, 0, 0, 0, 0, 64);
    // MLP temporal conv (GLU pre-activation), dil=1
    gemm(st, P[pb + 4], 0, 0, P[pb + 5], hcat, hcat, 1 << 30, y2, 0,
         64, 96, Kt, 1, (int)NT, T, 96 * NLh, (int)NLh, Lh,
         64 * NT, (int)NT, T, 0, 0, 0, 0, 0, 64);
    glu_kernel<<<GRID1(Bn * 32 * NT), 256, 0, st>>>(y2, g, NT, 64 * NT, 32 * NT, Bn * 32 * NT);
    // temporal attention
    at_f1_kernel<<<GRID1(Bn * T * Nn), 256, 0, st>>>(g, P[pb + 6], f1, Nn, T, Bn * (long long)T * Nn);
    at_f2_kernel<<<GRID1(Bn * 32 * T), 256, 0, st>>>(g, P[pb + 7], f2, Nn, T, Bn * 32LL * T);
    at_mid_kernel<<<GRID1(Bn * T * 32), 256, 0, st>>>(f1, P[pb + 8], mid, Nn, T, Bn * (long long)T * 32);
    at_logits_kernel<<<GRID1(Bn * T * T), 256, 0, st>>>(mid, f2, P[pb + 9], lg, T, Bn * (long long)T * T);
    at_v_kernel<<<GRID1(Bn * T * T), 256, 0, st>>>(P[pb + 10], lg, lv, T, Bn * (long long)T * T);
    at_bnstats_kernel<<<dim3(T), dim3(128), 0, st>>>(lv, sm + 64, 64, T);
    at_softmax_kernel<<<GRID1(Bn * T), 256, 0, st>>>(lv, sm + 64, coef, T, Bn * (long long)T);
    at_apply_kernel<<<GRID1(Bn * 32 * NT), 256, 0, st>>>(g, coef, res, tmp, Nn, T, Bn * 32 * NT);
    // BN2d -> out
    bn_stats_kernel<<<dim3(32), dim3(256), 0, st>>>(tmp, sm, 64, 32, NT);
    bn_apply_kernel<<<GRID1(Bn * 32 * NT), 256, 0, st>>>(tmp, outp, sm, 32, NT, Bn * 32 * NT);
}

// ---------------------------------------------------------------------------
extern "C" void kernel_launch(void* const* d_in, const int* in_sizes, int n_in,
                              void* d_out, int out_size, void* d_ws, size_t ws_size,
                              hipStream_t stream)
{
    if (n_in < 127) return;
    if (ws_size < (size_t)81000000 * 4) return;

    const float* IN    = (const float*)d_in[0];
    const float* INC   = (const float*)d_in[1];
    const float* INL   = (const float*)d_in[2];
    const float* TMASK = (const float*)d_in[3];
    const float* LMASK = (const float*)d_in[4];
    const float* P[122];
    for (int i = 0; i < 122; ++i) P[i] = (const float*)d_in[5 + i];

    float* out = (float*)d_out;
    float* Wf  = (float*)d_ws;

    // ---------------- workspace layout (floats) ----------------
    float* Ac   = Wf;                 // 4096
    float* Al   = Wf + 4096;          // 256
    float* SMa  = Wf + 4352;          // 1024 (input bn mv @0/8/16, gcn sm @128)
    float* CM   = Wf + 5376;          // 4096 colmean
    float* x0   = Wf + 16384;         // (B,32,512,12)
    float* xc0  = x0 + 12582912;      // (B,32,64,12)
    float* xl0  = xc0 + 1572864;      // (B,32,16,12)
    float* TRf1 = xl0 + 196608;
    float* TRf2 = TRf1 + 393216;
    float* TRg  = TRf2 + 131072;
    float* TRl  = TRg + 1048576;
    float* x1   = TRl + 2097152;      // (B,32,512,12)
    float* x4   = x1 + 12582912;      // (B,32,512,12)
    float* xg   = x4 + 12582912;      // (B,96,512,12), ends at 80,953,344

    float* Aout  = out + 3014656;     // A (512,512) output slice
    float* T3out = out + 393216;      // t3 (B,512,64)
    float* T6out = out + 2490368;     // t6 (B,512,16)

    // ---------------- phase A: input BN + start convs ----------------
    bn_stats_kernel<<<dim3(1), dim3(256), 0, stream>>>(IN, SMa, 64, 1, 512LL * 12);
    start_conv_kernel<<<GRID1(64LL * 32 * 512 * 12), 256, 0, stream>>>(IN, SMa, P[6], P[7], x0, 1, 512LL * 12, 64LL * 32 * 512 * 12);
    bn_stats_kernel<<<dim3(3), dim3(256), 0, stream>>>(INC, SMa + 8, 64, 3, 64LL * 12);
    start_conv_kernel<<<GRID1(64LL * 32 * 64 * 12), 256, 0, stream>>>(INC, SMa + 8, P[8], P[9], xc0, 3, 64LL * 12, 64LL * 32 * 64 * 12);
    bn_stats_kernel<<<dim3(2), dim3(256), 0, stream>>>(INL, SMa + 16, 64, 2, 16LL * 12);
    start_conv_kernel<<<GRID1(64LL * 32 * 16 * 12), 256, 0, stream>>>(INL, SMa + 16, P[10], P[11], xl0, 2, 16LL * 12, 64LL * 32 * 16 * 12);

    // ---------------- phase B: adaptive adjacency ----------------
    adaptive_kernel<<<dim3(512), dim3(256), 0, stream>>>(P[0], P[1], Aout, 512);
    adaptive_kernel<<<dim3(64), dim3(256), 0, stream>>>(P[2], P[3], Ac, 64);
    adaptive_kernel<<<dim3(16), dim3(256), 0, stream>>>(P[4], P[5], Al, 16);

    // ---------------- phase C: transmit 4 & 1, gate4 ----------------
    run_transmit(stream, x0, xl0, 16, 12, P[24], P[25], P[26], P[27], LMASK,
                 TRf1, TRf2, TRg, TRl, CM, TRl);
    gemm(stream, TRl, 512LL * 16, 0, 0, xl0, xl0, 1 << 30, x4, 0,
         512, 16, 1, 1, 32 * 12, 12, 32LL * 16 * 12, 12, 16 * 12,
         32LL * 512 * 12, 12, 512 * 12, 0, 0, 0, 0, 0, 64);
    run_transmit(stream, x0, xc0, 64, 12, P[12], P[13], P[14], P[15], TMASK,
                 TRf1, TRf2, TRg, TRl, CM, TRl);
    gemm(stream, TRl, 512LL * 64, 0, 0, xc0, xc0, 1 << 30, x1, 0,
         512, 64, 1, 1, 32 * 12, 12, 32LL * 64 * 12, 12, 64 * 12,
         32LL * 512 * 12, 12, 512 * 12, 0, 0, 0, 0, 0, 64);
    run_gate(stream, x0, x1, x4, P, 36, xg, 12);

    // ---------------- phase D: gcn_pool round 1 ----------------
    float* xB1   = x0;                        // (B,32,512,6)
    float* outC1 = xB1 + 6291456;             // (B,32,64,6)
    float* outL1 = outC1 + 786432;            // (B,32,16,6)
    run_gcn_pool(stream, xl0, 32, 16, 12, Al, P, 92, 3, outL1, x1, SMa + 128);
    run_gcn_pool(stream, xc0, 32, 64, 12, Ac, P, 70, 3, outC1, x1, SMa + 128);
    run_gcn_pool(stream, xg, 96, 512, 12, Aout, P, 48, 3, xB1, x1, SMa + 128);

    // ---------------- phase E: transmit 5 & 2, gate5, skip1 ----------------
    float* x5   = x1;
    float* x2b  = x5 + 6291456;
    float* xg5  = x2b + 6291456;              // (B,96,512,6)
    float* skip = xg5 + 18874368;             // (B,256,512,3)
    run_transmit(stream, xB1, outL1, 16, 6, P[28], P[29], P[30], P[31], LMASK,
                 TRf1, TRf2, TRg, TRl, CM, TRl);
    gemm(stream, TRl, 512LL * 16, 0, 0, outL1, outL1, 1 << 30, x5, 0,
         512, 16, 1, 1, 32 * 6, 6, 32LL * 16 * 6, 6, 16 * 6,
         32LL * 512 * 6, 6, 512 * 6, 0, 0, 0, 0, 0, 64);
    run_transmit(stream, xB1, outC1, 64, 6, P[16], P[17], P[18], P[19], TMASK,
                 TRf1, TRf2, TRg, TRl, CM, TRl);
    gemm(stream, TRl, 512LL * 64, 0, 0, outC1, outC1, 1 << 30, x2b, 0,
         512, 64, 1, 1, 32 * 6, 6, 32LL * 64 * 6, 6, 64 * 6,
         32LL * 512 * 6, 6, 512 * 6, 0, 0, 0, 0, 0, 64);
    run_gate(stream, xB1, x5, x2b, P, 40, xg5, 6);
    // skip1 on last-3 slice of gate5 output
    gemm(stream, P[114], 0, 0, P[115], xg5 + 3, xg5, 1 << 30, skip, 0,
         256, 96, 1, 1, 512 * 3, 3, 96LL * 512 * 6, 512 * 6, 6,
         256LL * 512 * 3, 512 * 3, 3, 0, 0, 0, 0, 0, 64);

    // ---------------- phase F: gcn_pool round 2 ----------------
    float* xB2   = x0;                        // (B,32,512,3)
    float* outL2 = xB2 + 3145728;             // (B,32,16,3)
    float* outC2 = outL2 + 98304;             // (B,32,64,3)
    float* scrF  = Wf + 7290880;
    run_gcn_pool(stream, outL1, 32, 16, 6, Al, P, 103, 2, outL2, scrF, SMa + 128);
    run_gcn_pool(stream, outC1, 32, 64, 6, Ac, P, 81, 2, outC2, scrF, SMa + 128);
    run_gcn_pool(stream, xg5, 96, 512, 6, Aout, P, 59, 2, xB2, scrF, SMa + 128);

    // ---------------- phase G: transmit 6 & 3 (outputs), gate6 ----------------
    float* x6   = outC2 + 393216;
    float* x3b  = x6 + 3145728;
    float* xg6  = x3b + 3145728;              // (B,96,512,3)
    float* endb = xg6 + 9437184;              // (B,512,512,1)
    run_transmit(stream, xB2, outL2, 16, 3, P[32], P[33], P[34], P[35], LMASK,
                 TRf1, TRf2, TRg, TRl, CM, T6out);
    gemm(stream, T6out, 512LL * 16, 0, 0, outL2, outL2, 1 << 30, x6, 0,
         512, 16, 1, 1, 32 * 3, 3, 32LL * 16 * 3, 3, 16 * 3,
         32LL * 512 * 3, 3, 512 * 3, 0, 0, 0, 0, 0, 64);
    run_transmit(stream, xB2, outC2, 64, 3, P[20], P[21], P[22], P[23], TMASK,
                 TRf1, TRf2, TRg, TRl, CM, T3out);
    gemm(stream, T3out, 512LL * 64, 0, 0, outC2, outC2, 1 << 30, x3b, 0,
         512, 64, 1, 1, 32 * 3, 3, 32LL * 64 * 3, 3, 64 * 3,
         32LL * 512 * 3, 3, 512 * 3, 0, 0, 0, 0, 0, 64);
    run_gate(stream, xB2, x6, x3b, P, 44, xg6, 3);

    // ---------------- phase H: skip2 (accumulate), end convs ----------------
    gemm(stream, P[116], 0, 0, P[117], xg6, xg6, 1 << 30, skip, 0,
         256, 96, 1, 1, 512 * 3, 3, 96LL * 512 * 3, 512 * 3, 3,
         256LL * 512 * 3, 512 * 3, 3, 0, 0, 0, 3, 0, 64);
    // end1: relu(skip) -> (1,3) conv -> relu
    gemm(stream, P[118], 0, 0, P[119], skip, skip, 1 << 30, endb, 0,
         512, 256, 3, 1, 512, 1, 256LL * 512 * 3, 512 * 3, 3,
         512LL * 512, 512, 1, 0, 0, 0, 1, 1, 64);
    // end2: 1x1 conv (12,512) -> output x slice
    gemm(stream, P[120], 0, 0, P[121], endb, endb, 1 << 30, out, 0,
         12, 512, 1, 1, 512, 1, 512LL * 512, 512, 1,
         12LL * 512, 512, 1, 0, 0, 0, 0, 0, 64);

    (void)in_sizes; (void)out_size;
}
